// JTNNEncoder_27934467293754
// MI455X (gfx1250) — matrix operand order repaired and checked by reading, not compiled, not run on hardware
//
#include <hip/hip_runtime.h>
#include <hip/hip_bf16.h>

typedef __attribute__((ext_vector_type(16))) __bf16 v16bf;
typedef __attribute__((ext_vector_type(8)))  __bf16 v8bf;
typedef __attribute__((ext_vector_type(8)))  float  v8f;

#define HDIM 512

__device__ __forceinline__ float sigmoidf(float x) { return 1.0f / (1.0f + __expf(-x)); }

// ---------------------------------------------------------------------------
// Re-tile a row-major (K x 512) f32 weight into the WMMA B-operand layout:
// per (kt,nt) 32x16 tile, lane l = hi*16+n holds 16 contiguous bf16 (K=16*hi+e).
// Each lane's fragment becomes one contiguous 32B load.
// ---------------------------------------------------------------------------
__global__ __launch_bounds__(256)
void k_tile_weight(const float* __restrict__ W, __bf16* __restrict__ Wt, int total) {
    int idx = blockIdx.x * 256 + threadIdx.x;
    if (idx >= total) return;
    int k  = idx >> 9;         // K row
    int n  = idx & 511;        // N col
    int kt = k >> 5, kk = k & 31;
    int nt = n >> 4, nn = n & 15;
    int hi = kk >> 4, e = kk & 15;
    size_t off = ((size_t)(kt * 32 + nt)) * 512 + (size_t)((hi * 16 + nn) * 16 + e);
    Wt[off] = (__bf16)W[idx];
}

// ---------------------------------------------------------------------------
// sx[e][c] = emb[wid[edge_src[e]]][c]  (bf16 activations for the GEMMs)
// ---------------------------------------------------------------------------
__global__ __launch_bounds__(256)
void k_build_sx(const int* __restrict__ wid, const int* __restrict__ edge_src,
                const float* __restrict__ emb, __bf16* __restrict__ sx, int total) {
    int idx = blockIdx.x * 256 + threadIdx.x;
    if (idx >= total) return;
    int e = idx >> 9, c = idx & 511;
    int node = edge_src[e];
    int w = wid[node];
    sx[idx] = (__bf16)emb[(size_t)w * HDIM + c];
}

// ---------------------------------------------------------------------------
// WMMA GEMM: out[M,512] = A[M,K](bf16) x Wt(K x 512, tiled bf16)
// flags: 1 = add bias, 2 = relu, 4 = bf16 output (else f32)
// Block = 128 threads = 4 waves; wave computes a 16x64 tile (4 WMMA accs,
// A fragment reused 4x). K-loop in steps of 32 with v_wmma_f32_16x16x32_bf16.
// M must be a multiple of 64 (all call sites guarantee this).
// ---------------------------------------------------------------------------
__global__ __launch_bounds__(128)
void k_gemm(const __bf16* __restrict__ A, int K,
            const __bf16* __restrict__ Wt,
            const float* __restrict__ bias,
            void* __restrict__ out, int flags) {
    const int wave = threadIdx.x >> 5;
    const int lane = threadIdx.x & 31;
    const int m0 = (blockIdx.x * 4 + wave) * 16;
    const int n0 = blockIdx.y * 64;
    const int mr = lane & 15;
    const int hi = lane >> 4;

    v8f acc[4] = {v8f{}, v8f{}, v8f{}, v8f{}};

    const __bf16* arow  = A + (size_t)(m0 + mr) * K;
    const __bf16* wbase = Wt + (size_t)(n0 >> 4) * 512 + (size_t)lane * 16;

    for (int kt = 0; kt < (K >> 5); ++kt) {
        // A fragment: lane (hi,m) holds K = {16hi..16hi+15} (matches tiled B layout)
        union { v16bf v; v8bf h[2]; } a;
        a.h[0] = *(const v8bf*)(arow + (kt << 5) + 16 * hi);
        a.h[1] = *(const v8bf*)(arow + (kt << 5) + 16 * hi + 8);

        const __bf16* wtk = wbase + (size_t)kt * 32 * 512;
        v16bf b0 = *(const v16bf*)(wtk);
        v16bf b1 = *(const v16bf*)(wtk + 512);
        v16bf b2 = *(const v16bf*)(wtk + 1024);
        v16bf b3 = *(const v16bf*)(wtk + 1536);

        acc[0] = __builtin_amdgcn_wmma_f32_16x16x32_bf16(false, a.v, false, b0, (short)0, acc[0], false, false);
        acc[1] = __builtin_amdgcn_wmma_f32_16x16x32_bf16(false, a.v, false, b1, (short)0, acc[1], false, false);
        acc[2] = __builtin_amdgcn_wmma_f32_16x16x32_bf16(false, a.v, false, b2, (short)0, acc[2], false, false);
        acc[3] = __builtin_amdgcn_wmma_f32_16x16x32_bf16(false, a.v, false, b3, (short)0, acc[3], false, false);
    }

    // C/D layout: lane = hi*16 + n ; VGPR v -> row m0 + 8*hi + v, col n
    const int n = lane & 15;
#pragma unroll
    for (int j = 0; j < 4; ++j) {
        int col = n0 + j * 16 + n;
        float bv = (flags & 1) ? bias[col] : 0.0f;
#pragma unroll
        for (int v = 0; v < 8; ++v) {
            int row = m0 + hi * 8 + v;
            float x = acc[j][v] + bv;
            if (flags & 2) x = x > 0.0f ? x : 0.0f;
            size_t o = (size_t)row * HDIM + col;
            if (flags & 4) ((__bf16*)out)[o] = (__bf16)x;
            else           ((float*)out)[o]  = x;
        }
    }
}

// ---------------------------------------------------------------------------
// Per-chunk gather: h_nei rows (bf16, GEMM A operand) + sum_h (f32 + bf16).
// preds row for idx==E is all-E, so padded entries produce zeros naturally.
// ---------------------------------------------------------------------------
__global__ __launch_bounds__(256)
void k_gather(const int* __restrict__ lvl, const int* __restrict__ preds, int P, int E,
              const float* __restrict__ h, __bf16* __restrict__ a_nei,
              float* __restrict__ sum_h_f, __bf16* __restrict__ sum_h_b) {
    int t = blockIdx.x * 256 + threadIdx.x;   // t < 2048*512
    int i = t >> 9, c = t & 511;
    int idx = lvl[i];
    const int* prow = preds + (size_t)idx * P;
    float s = 0.0f;
    for (int j = 0; j < P; ++j) {
        int p = prow[j];
        float v = (p >= E) ? 0.0f : h[(size_t)p * HDIM + c];
        a_nei[((size_t)(i * P + j)) * HDIM + c] = (__bf16)v;
        s += v;
    }
    sum_h_f[t] = s;
    sum_h_b[t] = (__bf16)s;
}

// ---------------------------------------------------------------------------
// r gate + gated sum: sum_g = sum_j sigmoid(pre_r + G_j + ur_b) * h_nei_j
// ---------------------------------------------------------------------------
__global__ __launch_bounds__(256)
void k_rgate(const int* __restrict__ lvl, const int* __restrict__ preds, int P, int E,
             const float* __restrict__ h, const __bf16* __restrict__ pre_r,
             const float* __restrict__ G, const float* __restrict__ ur_b,
             __bf16* __restrict__ sum_g_b) {
    int t = blockIdx.x * 256 + threadIdx.x;
    int i = t >> 9, c = t & 511;
    int idx = lvl[i];
    if (idx >= E) { sum_g_b[t] = (__bf16)0.0f; return; }
    float pr = (float)pre_r[(size_t)idx * HDIM + c];
    float ub = ur_b[c];
    const int* prow = preds + (size_t)idx * P;
    float g = 0.0f;
    for (int j = 0; j < P; ++j) {
        int p = prow[j];
        float hv = (p >= E) ? 0.0f : h[(size_t)p * HDIM + c];
        float r = sigmoidf(pr + G[((size_t)(i * P + j)) * HDIM + c] + ub);
        g += r * hv;
    }
    sum_g_b[t] = (__bf16)g;
}

// ---------------------------------------------------------------------------
// GRU combine: h[idx] = (1-z)*sum_h + z*tanh(...)
// ---------------------------------------------------------------------------
__global__ __launch_bounds__(256)
void k_combine(const int* __restrict__ lvl, int E,
               const float* __restrict__ zacc, const float* __restrict__ hacc,
               const float* __restrict__ pre_z, const float* __restrict__ pre_h,
               const float* __restrict__ sum_h_f, float* __restrict__ h) {
    int t = blockIdx.x * 256 + threadIdx.x;
    int i = t >> 9, c = t & 511;
    int idx = lvl[i];
    if (idx >= E) return;
    float z  = sigmoidf(zacc[t] + pre_z[(size_t)idx * HDIM + c]);
    float ph = tanhf(hacc[t] + pre_h[(size_t)idx * HDIM + c]);
    h[(size_t)idx * HDIM + c] = (1.0f - z) * sum_h_f[t] + z * ph;
}

// ---------------------------------------------------------------------------
// Root readout A operand: [x_root | sum(h[root_in])] in bf16 (2048 x 1024)
// ---------------------------------------------------------------------------
__global__ __launch_bounds__(256)
void k_root_prep(const int* __restrict__ wid, const int* __restrict__ root_nodes,
                 const int* __restrict__ root_in, int R,
                 const float* __restrict__ emb, const float* __restrict__ h,
                 __bf16* __restrict__ a_root, int total) {
    int t = blockIdx.x * 256 + threadIdx.x;
    if (t >= total) return;
    int b = t >> 10, c = t & 1023;
    float v;
    if (c < HDIM) {
        v = emb[(size_t)wid[root_nodes[b]] * HDIM + c];
    } else {
        v = 0.0f;
        int cc = c - HDIM;
        const int* rrow = root_in + (size_t)b * R;
        for (int r = 0; r < R; ++r) v += h[(size_t)rrow[r] * HDIM + cc];
    }
    a_root[t] = (__bf16)v;
}

extern "C" void kernel_launch(void* const* d_in, const int* in_sizes, int n_in,
                              void* d_out, int out_size, void* d_ws, size_t ws_size,
                              hipStream_t stream) {
    const int*   wid        = (const int*)d_in[0];
    const int*   edge_src   = (const int*)d_in[1];
    const int*   preds      = (const int*)d_in[2];
    const int*   levels     = (const int*)d_in[3];
    const int*   root_in    = (const int*)d_in[4];
    const int*   root_nodes = (const int*)d_in[5];
    const float* emb  = (const float*)d_in[6];
    const float* w_r  = (const float*)d_in[7];
    const float* ur_w = (const float*)d_in[8];
    const float* ur_b = (const float*)d_in[9];
    const float* wz_w = (const float*)d_in[10];
    const float* wz_b = (const float*)d_in[11];
    const float* wh_w = (const float*)d_in[12];
    const float* wh_b = (const float*)d_in[13];
    const float* wo_w = (const float*)d_in[14];
    const float* wo_b = (const float*)d_in[15];

    const int E  = in_sizes[1];
    const int P  = in_sizes[2] / (E + 1);
    const int LV = in_sizes[3];
    const int CH = 2048;                 // chunk = B entries; level rows are multiples of B,
    const int nchunks = LV / CH;         // so chunks never straddle a level boundary.
    const int R  = in_sizes[4] / 2048;

    float* h_out    = (float*)d_out;                        // h_edges (E x 512), written in-place
    float* root_out = (float*)d_out + (size_t)E * HDIM;     // root_vec (2048 x 512)

    char* wp = (char*)d_ws;
    auto alloc = [&](size_t bytes) -> char* {
        char* p = wp;
        wp += (bytes + 255) & ~(size_t)255;
        return p;
    };
    __bf16* wt_r    = (__bf16*)alloc((size_t)512 * 512 * 2);
    __bf16* wt_ur   = (__bf16*)alloc((size_t)512 * 512 * 2);
    __bf16* wt_z1   = (__bf16*)alloc((size_t)512 * 512 * 2);
    __bf16* wt_z2   = (__bf16*)alloc((size_t)512 * 512 * 2);
    __bf16* wt_h1   = (__bf16*)alloc((size_t)512 * 512 * 2);
    __bf16* wt_h2   = (__bf16*)alloc((size_t)512 * 512 * 2);
    __bf16* wt_o    = (__bf16*)alloc((size_t)1024 * 512 * 2);
    __bf16* sx      = (__bf16*)alloc((size_t)E * HDIM * 2);
    __bf16* pre_r   = (__bf16*)alloc((size_t)E * HDIM * 2);
    float*  pre_z   = (float*) alloc((size_t)E * HDIM * 4);
    float*  pre_h   = (float*) alloc((size_t)E * HDIM * 4);
    __bf16* a_nei   = (__bf16*)alloc((size_t)CH * P * HDIM * 2);
    float*  G       = (float*) alloc((size_t)CH * P * HDIM * 4);
    float*  sum_h_f = (float*) alloc((size_t)CH * HDIM * 4);
    __bf16* sum_h_b = (__bf16*)alloc((size_t)CH * HDIM * 2);
    __bf16* sum_g_b = (__bf16*)alloc((size_t)CH * HDIM * 2);
    float*  zacc    = (float*) alloc((size_t)CH * HDIM * 4);
    float*  hacc    = (float*) alloc((size_t)CH * HDIM * 4);
    __bf16* a_root  = (__bf16*)alloc((size_t)2048 * 1024 * 2);

    // ---- one-time weight retiling into WMMA B-operand layout (stays hot in L2) ----
    const int WG = (512 * 512 + 255) / 256;
    k_tile_weight<<<WG, 256, 0, stream>>>(w_r, wt_r, 512 * 512);
    k_tile_weight<<<WG, 256, 0, stream>>>(ur_w, wt_ur, 512 * 512);
    k_tile_weight<<<WG, 256, 0, stream>>>(wz_w, wt_z1, 512 * 512);
    k_tile_weight<<<WG, 256, 0, stream>>>(wz_w + 512 * 512, wt_z2, 512 * 512);
    k_tile_weight<<<WG, 256, 0, stream>>>(wh_w, wt_h1, 512 * 512);
    k_tile_weight<<<WG, 256, 0, stream>>>(wh_w + 512 * 512, wt_h2, 512 * 512);
    k_tile_weight<<<(1024 * 512 + 255) / 256, 256, 0, stream>>>(wo_w, wt_o, 1024 * 512);

    // ---- edge-independent precomputes (three big E x 512 x 512 GEMMs) ----
    const int totE = E * HDIM;
    k_build_sx<<<(totE + 255) / 256, 256, 0, stream>>>(wid, edge_src, emb, sx, totE);

    dim3 ge(E / 64, 8);
    k_gemm<<<ge, 128, 0, stream>>>(sx, 512, wt_r,  nullptr, pre_r, 4);   // bf16 out
    k_gemm<<<ge, 128, 0, stream>>>(sx, 512, wt_z1, wz_b,    pre_z, 1);   // + bias
    k_gemm<<<ge, 128, 0, stream>>>(sx, 512, wt_h1, wh_b,    pre_h, 1);   // + bias

    // ---- level schedule: sequential 2048-entry chunks ----
    const int TE = CH * HDIM;            // 1M elementwise threads per chunk
    for (int c = 0; c < nchunks; ++c) {
        const int* lvl = levels + (size_t)c * CH;
        k_gather<<<TE / 256, 256, 0, stream>>>(lvl, preds, P, E, h_out, a_nei, sum_h_f, sum_h_b);

        dim3 gn((CH * P) / 64, 8);
        k_gemm<<<gn, 128, 0, stream>>>(a_nei, 512, wt_ur, nullptr, G, 0);

        k_rgate<<<TE / 256, 256, 0, stream>>>(lvl, preds, P, E, h_out, pre_r, G, ur_b, sum_g_b);

        dim3 gc(CH / 64, 8);
        k_gemm<<<gc, 128, 0, stream>>>(sum_h_b, 512, wt_z2, nullptr, zacc, 0);
        k_gemm<<<gc, 128, 0, stream>>>(sum_g_b, 512, wt_h2, nullptr, hacc, 0);

        k_combine<<<TE / 256, 256, 0, stream>>>(lvl, E, zacc, hacc, pre_z, pre_h, sum_h_f, h_out);
    }

    // ---- root readout: relu([x_root | h_root] @ wo_w + wo_b) ----
    k_root_prep<<<(2048 * 1024) / 256, 256, 0, stream>>>(wid, root_nodes, root_in, R,
                                                         emb, h_out, a_root, 2048 * 1024);
    dim3 gr(2048 / 64, 8);
    k_gemm<<<gr, 128, 0, stream>>>(a_root, 1024, wt_o, wo_b, root_out, 1 | 2);
}
